// WDAModel_39599598469514
// MI455X (gfx1250) — matrix operand order repaired and checked by reading, not compiled, not run on hardware
//
#include <hip/hip_runtime.h>

typedef _Float16 f16;
typedef _Float16 v16h __attribute__((ext_vector_type(16)));
typedef _Float16 v8h  __attribute__((ext_vector_type(8)));
typedef float    v8f  __attribute__((ext_vector_type(8)));

union Frag16 { v16h v; v8h h[2]; };

__constant__ float c_DLO[8] = {-0.010597401784997278f, 0.032883011666982945f,
                               0.030841381835986965f, -0.18703481171888114f,
                               -0.02798376941698385f, 0.6308807679295904f,
                               0.7148465705525415f, 0.23037781330885523f};
__constant__ float c_DHI[8] = {-0.23037781330885523f, 0.7148465705525415f,
                               -0.6308807679295904f, -0.02798376941698385f,
                               0.18703481171888114f, 0.030841381835986965f,
                               -0.032883011666982945f, -0.010597401784997278f};
__constant__ float c_MEAN[3] = {0.48145466f, 0.4578275f, 0.40821073f};
__constant__ float c_STD[3]  = {0.26862954f, 0.26130258f, 0.27577711f};

__device__ __forceinline__ float blockReduceSum(float v, float* red) {
  int tid = threadIdx.x;
  red[tid] = v; __syncthreads();
  for (int s = 128; s > 0; s >>= 1) { if (tid < s) red[tid] += red[tid + s]; __syncthreads(); }
  float r = red[0]; __syncthreads();
  return r;
}

// ---------------- wavelet ----------------
__global__ void wda_scale_img(const float* __restrict__ x, float* __restrict__ xr, long total) {
  long i = (long)blockIdx.x * 256 + threadIdx.x; if (i >= total) return;
  int c = (int)((i / 50176) % 3);
  xr[i] = x[i] * c_STD[c] + c_MEAN[c];
}
__global__ void wda_renorm(float* __restrict__ xw, long total) {
  long i = (long)blockIdx.x * 256 + threadIdx.x; if (i >= total) return;
  int c = (int)((i / 50176) % 3);
  xw[i] = (xw[i] - c_MEAN[c]) / c_STD[c];
}
// DWT along one axis: in = [outer][n][inner], out ca/cd = [outer][m][inner]
__global__ void wda_dwt(const float* __restrict__ in, float* __restrict__ ca,
                        float* __restrict__ cd, long total, int n, int inner, int m) {
  long idx = (long)blockIdx.x * 256 + threadIdx.x; if (idx >= total) return;
  int ii = (int)(idx % inner); long t2 = idx / inner; int t = (int)(t2 % m); long o = t2 / m;
  const float* base = in + o * (long)n * inner + ii;
  float a = 0.f, d = 0.f;
  for (int j = 0; j < 8; ++j) {
    int s = 2 * t + 1 - j;
    if (s < 0) s = -s - 1; else if (s >= n) s = 2 * n - 1 - s;
    float v = base[(long)s * inner];
    a += v * c_DLO[j]; d += v * c_DHI[j];
  }
  ca[idx] = a; cd[idx] = d;
}
// IDWT along one axis: out = [outer][outL][inner], outL = 2l-6. ca may be a cropped view.
__global__ void wda_idwt(const float* __restrict__ ca, const float* __restrict__ cd,
                         float* __restrict__ out, long total, int outL, int inner, int l,
                         long ca_ost, int ca_nst, long cd_ost, int cd_nst) {
  long idx = (long)blockIdx.x * 256 + threadIdx.x; if (idx >= total) return;
  int ii = (int)(idx % inner); long t2 = idx / inner; int t = (int)(t2 % outL); long o = t2 / outL;
  const float* pa = ca + o * ca_ost + ii;
  const float* pd = cd + o * cd_ost + ii;
  float acc = 0.f;
  for (int k = 0; k < 8; ++k) {
    int m2 = t + k - 1;
    if (m2 >= 0 && m2 < 2 * l && ((m2 & 1) == 0)) {
      int s = m2 >> 1;
      acc += pa[(long)s * ca_nst] * c_DLO[k] + pd[(long)s * cd_nst] * c_DHI[k];
    }
  }
  out[idx] = acc;
}
// per-(b,c) exact median of |hh| via bit-bisection + variance -> BayesShrink threshold
__global__ __launch_bounds__(256) void wda_med_thr(const float* __restrict__ hh,
                                                   float* __restrict__ thr, int n) {
  __shared__ float red[256];
  __shared__ unsigned sLo0, sHi0, sLo1, sHi1;
  int bc = blockIdx.x; int tid = threadIdx.x;
  const float* p = hh + (long)bc * n;
  int k0 = (n - 1) / 2, k1 = n / 2;
  if (tid == 0) { sLo0 = 0u; sHi0 = 0x7F800000u; sLo1 = 0u; sHi1 = 0x7F800000u; }
  __syncthreads();
  for (int it = 0; it < 31; ++it) {
    unsigned lo0 = sLo0, hi0 = sHi0, lo1 = sLo1, hi1 = sHi1;
    unsigned mid0 = lo0 + ((hi0 - lo0) >> 1);
    unsigned mid1 = lo1 + ((hi1 - lo1) >> 1);
    float c0 = 0.f, c1 = 0.f;
    for (int i = tid; i < n; i += 256) {
      unsigned u = __float_as_uint(fabsf(p[i]));
      c0 += (u <= mid0) ? 1.f : 0.f;
      c1 += (u <= mid1) ? 1.f : 0.f;
    }
    float t0 = blockReduceSum(c0, red);
    float t1 = blockReduceSum(c1, red);
    if (tid == 0) {
      if (lo0 < hi0) { if ((int)t0 >= k0 + 1) sHi0 = mid0; else sLo0 = mid0 + 1; }
      if (lo1 < hi1) { if ((int)t1 >= k1 + 1) sHi1 = mid1; else sLo1 = mid1 + 1; }
    }
    __syncthreads();
  }
  float med = 0.5f * (__uint_as_float(sLo0) + __uint_as_float(sLo1));
  float s = 0.f, sq = 0.f;
  for (int i = tid; i < n; i += 256) { float v = p[i]; s += v; sq += v * v; }
  s = blockReduceSum(s, red);
  sq = blockReduceSum(sq, red);
  if (tid == 0) {
    float mean = s / n; float var = sq / n - mean * mean; if (var < 0.f) var = 0.f;
    float sigma = med / 0.6745f;
    float sx = sqrtf(fmaxf(var - sigma * sigma, 0.f));
    thr[bc] = sigma * sigma / fmaxf(sx, 1e-10f);
  }
}
__global__ void wda_soft(float* __restrict__ bnd, const float* __restrict__ thr,
                         int n, long total) {
  long i = (long)blockIdx.x * 256 + threadIdx.x; if (i >= total) return;
  float t = thr[i / n];
  float v = bnd[i];
  float a = fabsf(v) - t;
  bnd[i] = (a > 0.f) ? copysignf(a, v) : 0.f;
}

// ---------------- ViT helpers ----------------
// W[K][N] f32 -> Wt[N][Kpad] f16 (zero pad)
__global__ void wda_cvtT(const float* __restrict__ W, f16* __restrict__ O,
                         int K, int N, int Kpad, long total) {
  long i = (long)blockIdx.x * 256 + threadIdx.x; if (i >= total) return;
  int k = (int)(i % Kpad); long n = i / Kpad;
  O[i] = (k < K) ? (f16)W[(long)k * N + n] : (f16)0.f;
}
__global__ void wda_patches(const float* __restrict__ xw, f16* __restrict__ p, long total) {
  long i = (long)blockIdx.x * 256 + threadIdx.x; if (i >= total) return;
  int col = (int)(i % 608); long row = i / 608;
  f16 v = (f16)0.f;
  if (col < 588) {
    int b = (int)(row / 256), g = (int)(row % 256);
    int gy = g / 16, gx = g % 16;
    int c = col / 196, rem = col % 196, py = rem / 14, px = rem % 14;
    v = (f16)xw[(((long)(b * 3 + c) * 224) + gy * 14 + py) * 224 + gx * 14 + px];
  }
  p[i] = v;
}
__global__ void wda_assemble_h(const float* __restrict__ t, const float* __restrict__ patch_b,
                               const float* __restrict__ cls_tok, const float* __restrict__ pos,
                               float* __restrict__ h, long total) {
  long i = (long)blockIdx.x * 256 + threadIdx.x; if (i >= total) return;
  int c = (int)(i % 1024); long r = i / 1024; int s = (int)(r % 272); int b = (int)(r / 272);
  float v = 0.f;
  if (s == 0) v = cls_tok[c] + pos[c];
  else if (s <= 256) v = t[((long)b * 256 + (s - 1)) * 1024 + c] + patch_b[c] + pos[(long)s * 1024 + c];
  h[i] = v;
}
__global__ __launch_bounds__(256) void wda_ln(const float* __restrict__ X,
                                              const float* __restrict__ g,
                                              const float* __restrict__ b,
                                              f16* __restrict__ Y, int D) {
  long row = blockIdx.x;
  const float* x = X + row * D; f16* y = Y + row * D;
  __shared__ float red[256];
  int tid = threadIdx.x;
  float s = 0.f; for (int j = tid; j < D; j += 256) s += x[j];
  s = blockReduceSum(s, red); float m = s / D;
  float v = 0.f; for (int j = tid; j < D; j += 256) { float d = x[j] - m; v += d * d; }
  v = blockReduceSum(v, red); float r = rsqrtf(v / D + 1e-5f);
  for (int j = tid; j < D; j += 256) y[j] = (f16)((x[j] - m) * r * g[j] + b[j]);
}
__global__ void wda_qkv_split(const float* __restrict__ qkv, const float* __restrict__ bqkv,
                              f16* __restrict__ q, f16* __restrict__ k,
                              f16* __restrict__ vt, long total) {
  long i = (long)blockIdx.x * 256 + threadIdx.x; if (i >= total) return;
  int j = (int)(i % 3072); long r = i / 3072; int s = (int)(r % 272); int b = (int)(r / 272);
  float v = qkv[i] + bqkv[j];
  int t = j / 1024, jj = j % 1024, h = jj / 64, d = jj % 64;
  long bh = (long)(b * 16 + h);
  if (t == 0)      q[(bh * 272 + s) * 64 + d] = (f16)v;
  else if (t == 1) k[(bh * 272 + s) * 64 + d] = (f16)v;
  else             vt[(bh * 64 + d) * 272 + s] = (f16)v;
}
__global__ __launch_bounds__(256) void wda_softmax(const float* __restrict__ S,
                                                   f16* __restrict__ O,
                                                   int ld, int valid, float scale) {
  long row = blockIdx.x;
  const float* p = S + row * ld; f16* o = O + row * ld;
  __shared__ float red[256];
  int tid = threadIdx.x;
  float mx = -1e30f;
  for (int j = tid; j < valid; j += 256) mx = fmaxf(mx, p[j] * scale);
  red[tid] = mx; __syncthreads();
  for (int s = 128; s > 0; s >>= 1) { if (tid < s) red[tid] = fmaxf(red[tid], red[tid + s]); __syncthreads(); }
  mx = red[0]; __syncthreads();
  float sum = 0.f;
  for (int j = tid; j < valid; j += 256) sum += expf(p[j] * scale - mx);
  sum = blockReduceSum(sum, red);
  float inv = 1.f / sum;
  for (int j = tid; j < ld; j += 256)
    o[j] = (j < valid) ? (f16)(expf(p[j] * scale - mx) * inv) : (f16)0.f;
}
__global__ void wda_merge_o(const float* __restrict__ ob, f16* __restrict__ o16, long total) {
  long i = (long)blockIdx.x * 256 + threadIdx.x; if (i >= total) return;
  int j = (int)(i % 1024); long r = i / 1024; int s = (int)(r % 272); int b = (int)(r / 272);
  int h = j / 64, d = j % 64;
  o16[i] = (f16)ob[((long)(b * 16 + h) * 272 + s) * 64 + d];
}
__global__ void wda_resid(float* __restrict__ h, const float* __restrict__ t,
                          const float* __restrict__ bias, int C, long total) {
  long i = (long)blockIdx.x * 256 + threadIdx.x; if (i >= total) return;
  h[i] += t[i] + bias[i % C];
}
__global__ void wda_silu(const float* __restrict__ t, const float* __restrict__ b1m,
                         f16* __restrict__ u, long total) {
  long i = (long)blockIdx.x * 256 + threadIdx.x; if (i >= total) return;
  float v = t[i] + b1m[i % 4096];
  u[i] = (f16)(v / (1.f + expf(-1.702f * v)));
}
__global__ __launch_bounds__(256) void wda_cls_ln(const float* __restrict__ h,
                                                  const float* __restrict__ g,
                                                  const float* __restrict__ b,
                                                  f16* __restrict__ cls, int D) {
  int bb = blockIdx.x; int tid = threadIdx.x;
  __shared__ float red[256];
  if (bb >= 8) { for (int j = tid; j < D; j += 256) cls[(long)bb * D + j] = (f16)0.f; return; }
  const float* x = h + (long)bb * 272 * D;
  float s = 0.f; for (int j = tid; j < D; j += 256) s += x[j];
  s = blockReduceSum(s, red); float m = s / D;
  float v = 0.f; for (int j = tid; j < D; j += 256) { float d = x[j] - m; v += d * d; }
  v = blockReduceSum(v, red); float r = rsqrtf(v / D + 1e-5f);
  for (int j = tid; j < D; j += 256) cls[(long)bb * D + j] = (f16)((x[j] - m) * r * g[j] + b[j]);
}
__global__ __launch_bounds__(256) void wda_final(const float* __restrict__ feat,
                                                 const float* __restrict__ bn_g,
                                                 const float* __restrict__ bn_b,
                                                 const float* __restrict__ cls_w,
                                                 const float* __restrict__ cls_b,
                                                 float* __restrict__ out) {
  __shared__ float red[256];
  int tid = threadIdx.x;
  float acc[8]; for (int i = 0; i < 8; ++i) acc[i] = 0.f;
  for (int j = tid; j < 768; j += 256) {
    float f[8]; float mu = 0.f;
    for (int i = 0; i < 8; ++i) { f[i] = feat[(long)i * 768 + j]; mu += f[i]; }
    mu *= 0.125f;
    float var = 0.f; for (int i = 0; i < 8; ++i) { float d = f[i] - mu; var += d * d; }
    var *= 0.125f;
    float r = rsqrtf(var + 1e-5f);
    float gg = bn_g[j], bb = bn_b[j], w = cls_w[j];
    for (int i = 0; i < 8; ++i) {
      float fn = (f[i] - mu) * r * gg + bb;
      acc[i] += fn * w;
      out[8 + (long)i * 768 + j] = f[i];
    }
  }
  for (int i = 0; i < 8; ++i) {
    float s = blockReduceSum(acc[i], red);
    if (tid == 0) out[i] = s + cls_b[0];
  }
}

// ---------------- WMMA GEMM: C[M,N] = A[M,K] * Bt[N,K]^T, f16 in, f32 out ----------------
// Double-buffered LDS; full tiles staged via gfx1250 global_load_async_to_lds_b128
// (ASYNCcnt-tracked DMA), edge tiles via uniform-branch guarded ds stores.
constexpr int BM = 64, BN = 128, BK = 32;

__device__ __forceinline__ void wda_stage(
    const f16* __restrict__ A, const f16* __restrict__ Bt,
    f16 (* __restrict__ As)[BK + 8], f16 (* __restrict__ Bs)[BK + 8],
    int M, int N, int K, int lda, int ldb,
    int bm, int bn, int k0, int tid, bool fullM, bool fullN) {
  const bool fullK = (k0 + BK <= K);
  const int ar = tid >> 2;
  const int akc = (tid & 3) * 8;
  if (fullM && fullK) {  // uniform branch: pure async DMA path
    const f16* src = A + (long)(bm + ar) * lda + (k0 + akc);
    unsigned dst = (unsigned)(size_t)&As[ar][akc];  // LDS addr = low 32 bits of generic ptr
    asm volatile("global_load_async_to_lds_b128 %0, %1, off"
                 :: "v"(dst), "v"((unsigned long long)(size_t)src) : "memory");
  } else {
    const int gr = bm + ar, gk = k0 + akc;
    union { int4 i4; f16 h[8]; } t;
    const f16* src = A + (long)gr * lda + gk;
    for (int q = 0; q < 8; ++q)
      t.h[q] = (gr < M && gk + q < K) ? src[q] : (f16)0.f;
    *(int4*)&As[ar][akc] = t.i4;
  }
  for (int c = 0; c < 2; ++c) {
    const int chunk = tid * 2 + c;
    const int bnr = chunk >> 2;
    const int bkc = (chunk & 3) * 8;
    if (fullN && fullK) {  // uniform branch
      const f16* src = Bt + (long)(bn + bnr) * ldb + (k0 + bkc);
      unsigned dst = (unsigned)(size_t)&Bs[bnr][bkc];
      asm volatile("global_load_async_to_lds_b128 %0, %1, off"
                   :: "v"(dst), "v"((unsigned long long)(size_t)src) : "memory");
    } else {
      const int gn = bn + bnr, gk = k0 + bkc;
      union { int4 i4; f16 h[8]; } t;
      const f16* src = Bt + (long)gn * ldb + gk;
      for (int q = 0; q < 8; ++q)
        t.h[q] = (gn < N && gk + q < K) ? src[q] : (f16)0.f;
      *(int4*)&Bs[bnr][bkc] = t.i4;
    }
  }
}

__global__ __launch_bounds__(256) void wda_gemm(
    const f16* __restrict__ A, const f16* __restrict__ Bt, float* __restrict__ C,
    int M, int N, int K, int lda, int ldb, int ldc,
    long sA, long sB, long sC) {
  __shared__ __align__(16) f16 As[2][BM][BK + 8];
  __shared__ __align__(16) f16 Bs[2][BN][BK + 8];
  const int z = blockIdx.z;
  A  += (long)z * sA; Bt += (long)z * sB; C += (long)z * sC;
  const int bm = blockIdx.y * BM;
  const int bn = blockIdx.x * BN;
  const int tid = threadIdx.x;
  const int lane = tid & 31;
  const int w = tid >> 5;
  const int wm = (w >> 2) * 32;     // 2 wave rows
  const int wn = (w & 3) * 32;      // 4 wave cols
  const int l16 = lane & 15;
  const int lh = lane >> 4;
  const bool fullM = (bm + BM <= M);
  const bool fullN = (bn + BN <= N);
  v8f acc[2][2] = {};
  const int kT = (K + BK - 1) / BK;
  // prologue: stage tile 0 into buffer 0
  wda_stage(A, Bt, As[0], Bs[0], M, N, K, lda, ldb, bm, bn, 0, tid, fullM, fullN);
  for (int kt = 0; kt < kT; ++kt) {
    asm volatile("s_wait_asynccnt 0x0" ::: "memory");
    __syncthreads();
    const int cb = kt & 1;
    if (kt + 1 < kT)  // pipeline: async-stage next tile into the other buffer
      wda_stage(A, Bt, As[cb ^ 1], Bs[cb ^ 1], M, N, K, lda, ldb,
                bm, bn, (kt + 1) * BK, tid, fullM, fullN);
    // fragments per ISA 16-bit A(16x32)/B(32x16) wave32 layouts
    Frag16 af[2], bf[2];
    for (int mi = 0; mi < 2; ++mi) {
      const int row = wm + mi * 16 + l16;
      const int kb = lh * 8;
      af[mi].h[0] = *(const v8h*)&As[cb][row][kb];
      af[mi].h[1] = *(const v8h*)&As[cb][row][kb + 16];
    }
    for (int ni = 0; ni < 2; ++ni) {
      const int col = wn + ni * 16 + l16;
      const int kb = lh * 8;
      bf[ni].h[0] = *(const v8h*)&Bs[cb][col][kb];
      bf[ni].h[1] = *(const v8h*)&Bs[cb][col][kb + 16];
    }
    for (int mi = 0; mi < 2; ++mi)
      for (int ni = 0; ni < 2; ++ni)
        acc[mi][ni] = __builtin_amdgcn_wmma_f32_16x16x32_f16(
            false, af[mi].v, false, bf[ni].v, (short)0, acc[mi][ni], false, false);
  }
  if (fullM && fullN) {  // uniform fast store
    for (int mi = 0; mi < 2; ++mi) {
      const int grb = bm + wm + mi * 16 + lh * 8;
      for (int ni = 0; ni < 2; ++ni) {
        const int gc = bn + wn + ni * 16 + l16;
        for (int r = 0; r < 8; ++r)
          C[(long)(grb + r) * ldc + gc] = acc[mi][ni][r];
      }
    }
  } else {
    for (int mi = 0; mi < 2; ++mi) {
      const int grb = bm + wm + mi * 16 + lh * 8;
      for (int ni = 0; ni < 2; ++ni) {
        const int gc = bn + wn + ni * 16 + l16;
        if (gc < N) {
          for (int r = 0; r < 8; ++r) {
            const int gr = grb + r;
            if (gr < M) C[(long)gr * ldc + gc] = acc[mi][ni][r];
          }
        }
      }
    }
  }
}

// ---------------- host orchestration ----------------
extern "C" void kernel_launch(void* const* d_in, const int* in_sizes, int n_in,
                              void* d_out, int out_size, void* d_ws, size_t ws_size,
                              hipStream_t stream) {
  (void)in_sizes; (void)n_in; (void)out_size; (void)ws_size;
  const float* x        = (const float*)d_in[0];
  const float* patch_w  = (const float*)d_in[1];
  const float* patch_b  = (const float*)d_in[2];
  const float* cls_tok  = (const float*)d_in[3];
  const float* pos_emb  = (const float*)d_in[4];
  const float* g1   = (const float*)d_in[5];
  const float* b1   = (const float*)d_in[6];
  const float* wqkv = (const float*)d_in[7];
  const float* bqkv = (const float*)d_in[8];
  const float* wo   = (const float*)d_in[9];
  const float* bo   = (const float*)d_in[10];
  const float* g2   = (const float*)d_in[11];
  const float* b2   = (const float*)d_in[12];
  const float* w1   = (const float*)d_in[13];
  const float* b1m  = (const float*)d_in[14];
  const float* w2   = (const float*)d_in[15];
  const float* b2m  = (const float*)d_in[16];
  const float* lnf_g = (const float*)d_in[17];
  const float* lnf_b = (const float*)d_in[18];
  const float* vis_proj = (const float*)d_in[19];
  const float* bn_g  = (const float*)d_in[20];
  const float* bn_b  = (const float*)d_in[21];
  const float* cls_w = (const float*)d_in[22];
  const float* cls_b = (const float*)d_in[23];
  float* out = (float*)d_out;

  char* cur = (char*)d_ws;
  auto alloc = [&](size_t bytes) -> void* {
    void* p = (void*)cur; cur += (bytes + 255) & ~(size_t)255; return p;
  };
  auto gdiv = [](long a, long b) -> unsigned { return (unsigned)((a + b - 1) / b); };

  // wavelet buffers (B*3 = 24 slabs)
  float* xr   = (float*)alloc(24L * 224 * 224 * 4);
  float* lo1  = (float*)alloc(24L * 224 * 115 * 4);
  float* hi1  = (float*)alloc(24L * 224 * 115 * 4);
  float* ca1  = (float*)alloc(24L * 115 * 115 * 4);
  float* lh1  = (float*)alloc(24L * 115 * 115 * 4);
  float* hl1  = (float*)alloc(24L * 115 * 115 * 4);
  float* hh1  = (float*)alloc(24L * 115 * 115 * 4);
  float* lo2  = (float*)alloc(24L * 115 * 61 * 4);
  float* hi2  = (float*)alloc(24L * 115 * 61 * 4);
  float* ca2  = (float*)alloc(24L * 61 * 61 * 4);
  float* lh2  = (float*)alloc(24L * 61 * 61 * 4);
  float* hl2  = (float*)alloc(24L * 61 * 61 * 4);
  float* hh2  = (float*)alloc(24L * 61 * 61 * 4);
  float* lo3  = (float*)alloc(24L * 61 * 34 * 4);
  float* hi3  = (float*)alloc(24L * 61 * 34 * 4);
  float* ca3  = (float*)alloc(24L * 34 * 34 * 4);
  float* lh3  = (float*)alloc(24L * 34 * 34 * 4);
  float* hl3  = (float*)alloc(24L * 34 * 34 * 4);
  float* hh3  = (float*)alloc(24L * 34 * 34 * 4);
  float* thr1 = (float*)alloc(24 * 4);
  float* thr2 = (float*)alloc(24 * 4);
  float* thr3 = (float*)alloc(24 * 4);
  float* r3lo = (float*)alloc(24L * 62 * 34 * 4);
  float* r3hi = (float*)alloc(24L * 62 * 34 * 4);
  float* caR2 = (float*)alloc(24L * 62 * 62 * 4);
  float* r2lo = (float*)alloc(24L * 116 * 61 * 4);
  float* r2hi = (float*)alloc(24L * 116 * 61 * 4);
  float* caR1 = (float*)alloc(24L * 116 * 116 * 4);
  float* r1lo = (float*)alloc(24L * 224 * 115 * 4);
  float* r1hi = (float*)alloc(24L * 224 * 115 * 4);

  // ViT buffers
  f16* p16      = (f16*)alloc(2048L * 608 * 2);
  f16* wpatchT  = (f16*)alloc(1024L * 608 * 2);
  f16* visprojT = (f16*)alloc(768L * 1024 * 2);
  f16* wqkvT    = (f16*)alloc(3072L * 1024 * 2);
  f16* woT      = (f16*)alloc(1024L * 1024 * 2);
  f16* w1T      = (f16*)alloc(4096L * 1024 * 2);
  f16* w2T      = (f16*)alloc(1024L * 4096 * 2);
  float* tmp    = (float*)alloc(2176L * 4096 * 4);
  float* hbuf   = (float*)alloc(2176L * 1024 * 4);
  f16* y16      = (f16*)alloc(2176L * 1024 * 2);
  f16* q16      = (f16*)alloc(128L * 272 * 64 * 2);
  f16* k16      = (f16*)alloc(128L * 272 * 64 * 2);
  f16* v16t     = (f16*)alloc(128L * 64 * 272 * 2);
  float* att    = (float*)alloc(128L * 272 * 272 * 4);
  f16* att16    = (f16*)alloc(128L * 272 * 272 * 2);
  float* obuf   = (float*)alloc(128L * 272 * 64 * 4);
  f16* o16      = (f16*)alloc(2176L * 1024 * 2);
  f16* u16      = (f16*)alloc(2176L * 4096 * 2);
  f16* cls16    = (f16*)alloc(16L * 1024 * 2);
  float* featb  = (float*)alloc(16L * 768 * 4);

  auto gemm = [&](const f16* A, const f16* Bt, float* C, int M, int N, int K,
                  int lda, int ldb, int ldc, long sA, long sB, long sC, int batch) {
    dim3 g((N + BN - 1) / BN, (M + BM - 1) / BM, batch);
    wda_gemm<<<g, 256, 0, stream>>>(A, Bt, C, M, N, K, lda, ldb, ldc, sA, sB, sC);
  };

  // ---- wavelet denoise ----
  long nimg = 24L * 224 * 224;
  wda_scale_img<<<gdiv(nimg, 256), 256, 0, stream>>>(x, xr, nimg);
  // level 1
  wda_dwt<<<gdiv(24L*224*115,256),256,0,stream>>>(xr, lo1, hi1, 24L*224*115, 224, 1, 115);
  wda_dwt<<<gdiv(24L*115*115,256),256,0,stream>>>(lo1, ca1, lh1, 24L*115*115, 224, 115, 115);
  wda_dwt<<<gdiv(24L*115*115,256),256,0,stream>>>(hi1, hl1, hh1, 24L*115*115, 224, 115, 115);
  // level 2
  wda_dwt<<<gdiv(24L*115*61,256),256,0,stream>>>(ca1, lo2, hi2, 24L*115*61, 115, 1, 61);
  wda_dwt<<<gdiv(24L*61*61,256),256,0,stream>>>(lo2, ca2, lh2, 24L*61*61, 115, 61, 61);
  wda_dwt<<<gdiv(24L*61*61,256),256,0,stream>>>(hi2, hl2, hh2, 24L*61*61, 115, 61, 61);
  // level 3
  wda_dwt<<<gdiv(24L*61*34,256),256,0,stream>>>(ca2, lo3, hi3, 24L*61*34, 61, 1, 34);
  wda_dwt<<<gdiv(24L*34*34,256),256,0,stream>>>(lo3, ca3, lh3, 24L*34*34, 61, 34, 34);
  wda_dwt<<<gdiv(24L*34*34,256),256,0,stream>>>(hi3, hl3, hh3, 24L*34*34, 61, 34, 34);
  // thresholds + soft shrink
  wda_med_thr<<<24, 256, 0, stream>>>(hh1, thr1, 115 * 115);
  wda_med_thr<<<24, 256, 0, stream>>>(hh2, thr2, 61 * 61);
  wda_med_thr<<<24, 256, 0, stream>>>(hh3, thr3, 34 * 34);
  wda_soft<<<gdiv(24L*115*115,256),256,0,stream>>>(lh1, thr1, 115*115, 24L*115*115);
  wda_soft<<<gdiv(24L*115*115,256),256,0,stream>>>(hl1, thr1, 115*115, 24L*115*115);
  wda_soft<<<gdiv(24L*115*115,256),256,0,stream>>>(hh1, thr1, 115*115, 24L*115*115);
  wda_soft<<<gdiv(24L*61*61,256),256,0,stream>>>(lh2, thr2, 61*61, 24L*61*61);
  wda_soft<<<gdiv(24L*61*61,256),256,0,stream>>>(hl2, thr2, 61*61, 24L*61*61);
  wda_soft<<<gdiv(24L*61*61,256),256,0,stream>>>(hh2, thr2, 61*61, 24L*61*61);
  wda_soft<<<gdiv(24L*34*34,256),256,0,stream>>>(lh3, thr3, 34*34, 24L*34*34);
  wda_soft<<<gdiv(24L*34*34,256),256,0,stream>>>(hl3, thr3, 34*34, 24L*34*34);
  wda_soft<<<gdiv(24L*34*34,256),256,0,stream>>>(hh3, thr3, 34*34, 24L*34*34);
  // reconstruct level 3: 34 -> 62
  wda_idwt<<<gdiv(24L*62*34,256),256,0,stream>>>(ca3, lh3, r3lo, 24L*62*34, 62, 34, 34,
                                                 34L*34, 34, 34L*34, 34);
  wda_idwt<<<gdiv(24L*62*34,256),256,0,stream>>>(hl3, hh3, r3hi, 24L*62*34, 62, 34, 34,
                                                 34L*34, 34, 34L*34, 34);
  wda_idwt<<<gdiv(24L*62*62,256),256,0,stream>>>(r3lo, r3hi, caR2, 24L*62*62, 62, 1, 34,
                                                 34L, 1, 34L, 1);
  // level 2: crop 62->61, 61 -> 116
  wda_idwt<<<gdiv(24L*116*61,256),256,0,stream>>>(caR2, lh2, r2lo, 24L*116*61, 116, 61, 61,
                                                  62L*62, 62, 61L*61, 61);
  wda_idwt<<<gdiv(24L*116*61,256),256,0,stream>>>(hl2, hh2, r2hi, 24L*116*61, 116, 61, 61,
                                                  61L*61, 61, 61L*61, 61);
  wda_idwt<<<gdiv(24L*116*116,256),256,0,stream>>>(r2lo, r2hi, caR1, 24L*116*116, 116, 1, 61,
                                                   61L, 1, 61L, 1);
  // level 1: crop 116->115, 115 -> 224 (reuse xr as output)
  wda_idwt<<<gdiv(24L*224*115,256),256,0,stream>>>(caR1, lh1, r1lo, 24L*224*115, 224, 115, 115,
                                                   116L*116, 116, 115L*115, 115);
  wda_idwt<<<gdiv(24L*224*115,256),256,0,stream>>>(hl1, hh1, r1hi, 24L*224*115, 224, 115, 115,
                                                   115L*115, 115, 115L*115, 115);
  wda_idwt<<<gdiv(24L*224*224,256),256,0,stream>>>(r1lo, r1hi, xr, 24L*224*224, 224, 1, 115,
                                                   115L, 1, 115L, 1);
  wda_renorm<<<gdiv(nimg, 256), 256, 0, stream>>>(xr, nimg);

  // ---- ViT ----
  wda_cvtT<<<gdiv(1024L*608,256),256,0,stream>>>(patch_w, wpatchT, 588, 1024, 608, 1024L*608);
  wda_cvtT<<<gdiv(768L*1024,256),256,0,stream>>>(vis_proj, visprojT, 1024, 768, 1024, 768L*1024);
  wda_patches<<<gdiv(2048L*608,256),256,0,stream>>>(xr, p16, 2048L*608);
  gemm(p16, wpatchT, tmp, 2048, 1024, 608, 608, 608, 1024, 0, 0, 0, 1);
  wda_assemble_h<<<gdiv(2176L*1024,256),256,0,stream>>>(tmp, patch_b, cls_tok, pos_emb,
                                                        hbuf, 2176L*1024);
  for (int l = 0; l < 6; ++l) {
    wda_cvtT<<<gdiv(3072L*1024,256),256,0,stream>>>(wqkv + (long)l*1024*3072, wqkvT,
                                                    1024, 3072, 1024, 3072L*1024);
    wda_cvtT<<<gdiv(1024L*1024,256),256,0,stream>>>(wo + (long)l*1024*1024, woT,
                                                    1024, 1024, 1024, 1024L*1024);
    wda_cvtT<<<gdiv(4096L*1024,256),256,0,stream>>>(w1 + (long)l*1024*4096, w1T,
                                                    1024, 4096, 1024, 4096L*1024);
    wda_cvtT<<<gdiv(1024L*4096,256),256,0,stream>>>(w2 + (long)l*4096*1024, w2T,
                                                    4096, 1024, 4096, 1024L*4096);
    // attention
    wda_ln<<<2176, 256, 0, stream>>>(hbuf, g1 + l*1024, b1 + l*1024, y16, 1024);
    gemm(y16, wqkvT, tmp, 2176, 3072, 1024, 1024, 1024, 3072, 0, 0, 0, 1);
    wda_qkv_split<<<gdiv(2176L*3072,256),256,0,stream>>>(tmp, bqkv + l*3072,
                                                         q16, k16, v16t, 2176L*3072);
    gemm(q16, k16, att, 272, 272, 64, 64, 64, 272,
         272L*64, 272L*64, 272L*272, 128);
    wda_softmax<<<128*272, 256, 0, stream>>>(att, att16, 272, 257, 0.125f);
    gemm(att16, v16t, obuf, 272, 64, 272, 272, 272, 64,
         272L*272, 64L*272, 272L*64, 128);
    wda_merge_o<<<gdiv(2176L*1024,256),256,0,stream>>>(obuf, o16, 2176L*1024);
    gemm(o16, woT, tmp, 2176, 1024, 1024, 1024, 1024, 1024, 0, 0, 0, 1);
    wda_resid<<<gdiv(2176L*1024,256),256,0,stream>>>(hbuf, tmp, bo + l*1024, 1024, 2176L*1024);
    // MLP
    wda_ln<<<2176, 256, 0, stream>>>(hbuf, g2 + l*1024, b2 + l*1024, y16, 1024);
    gemm(y16, w1T, tmp, 2176, 4096, 1024, 1024, 1024, 4096, 0, 0, 0, 1);
    wda_silu<<<gdiv(2176L*4096,256),256,0,stream>>>(tmp, b1m + l*4096, u16, 2176L*4096);
    gemm(u16, w2T, tmp, 2176, 1024, 4096, 4096, 4096, 1024, 0, 0, 0, 1);
    wda_resid<<<gdiv(2176L*1024,256),256,0,stream>>>(hbuf, tmp, b2m + l*1024, 1024, 2176L*1024);
  }
  wda_cls_ln<<<16, 256, 0, stream>>>(hbuf, lnf_g, lnf_b, cls16, 1024);
  gemm(cls16, visprojT, featb, 16, 768, 1024, 1024, 1024, 768, 0, 0, 0, 1);
  wda_final<<<1, 256, 0, stream>>>(featb, bn_g, bn_b, cls_w, cls_b, out);
}
